// DotProductAttention_56272661512694
// MI455X (gfx1250) — matrix operand order repaired and checked by reading, not compiled, not run on hardware
//
#include <hip/hip_runtime.h>
#include <math.h>

// ---------------------------------------------------------------------------
// Flash attention for BH=16, S=2048, D=128 (fp32 in/out, bf16 WMMA compute)
// v3: DPP16 butterfly reductions (no ds_bpermute), 2 barriers/iter,
//     block-cooperative bf16 LDS staging, NT mask/output streaming.
// ---------------------------------------------------------------------------

typedef __bf16 bf16;
typedef __attribute__((ext_vector_type(16))) __bf16 v16bf;
typedef __attribute__((ext_vector_type(8)))  __bf16 v8bf;
typedef __attribute__((ext_vector_type(2)))  __bf16 v2bf;
typedef __attribute__((ext_vector_type(8)))  float  v8f;

#define SEQ   2048
#define DIM   128
#define NBH   16
#define BN    32      // kv rows per inner iteration
#define WAVES 4       // 4 waves * 16 q-rows = 64 q rows per workgroup

// padded LDS strides (bf16 elements) to spread 16-lane reads across banks
#define KT_STRIDE 136   // K tile rows: 32 x 128 (rows 272B apart, 16B-aligned)
#define VT_STRIDE 40    // V^T rows (d): 128 x 32 (rows 80B apart, 16B-aligned)
#define PT_STRIDE 40    // P rows: 16 x 32 per wave

__device__ __forceinline__ v8bf cvt8(float4 a, float4 b) {
  v8bf w;
  w[0]=(bf16)a.x; w[1]=(bf16)a.y; w[2]=(bf16)a.z; w[3]=(bf16)a.w;
  w[4]=(bf16)b.x; w[5]=(bf16)b.y; w[6]=(bf16)b.z; w[7]=(bf16)b.w;
  return w;
}

// Assemble a 16-element bf16 fragment from two contiguous 16B LDS reads.
__device__ __forceinline__ v16bf ldsfrag16(const __bf16* p) {
  v8bf lo = *(const v8bf*)p;
  v8bf hi = *(const v8bf*)(p + 8);
  return __builtin_shufflevector(lo, hi, 0,1,2,3,4,5,6,7,8,9,10,11,12,13,14,15);
}

// A-fragment (two 8-runs per lane-half, ISA 7.12.2) from two 16B LDS reads.
__device__ __forceinline__ v16bf ldsfragA(const __bf16* row, int h) {
  v8bf lo = *(const v8bf*)(row + 8 * h);
  v8bf hi = *(const v8bf*)(row + 16 + 8 * h);
  return __builtin_shufflevector(lo, hi, 0,1,2,3,4,5,6,7,8,9,10,11,12,13,14,15);
}

// A-fragment of Q directly from global fp32.
__device__ __forceinline__ v16bf load_fragA_f32(const float* row, int h) {
  const float* p = row + 8 * h;
  float4 a = *(const float4*)(p + 0);
  float4 b = *(const float4*)(p + 4);
  float4 c = *(const float4*)(p + 16);
  float4 d = *(const float4*)(p + 20);
  v8bf lo = cvt8(a, b);
  v8bf hi = cvt8(c, d);
  return __builtin_shufflevector(lo, hi, 0,1,2,3,4,5,6,7,8,9,10,11,12,13,14,15);
}

// ---- DPP16 lane-swizzles: pure-VALU 16-lane butterfly (no LDS traffic) ----
// quad_perm(1,0,3,2)=0xB1 (xor1), quad_perm(2,3,0,1)=0x4E (xor2),
// row_ror:4=0x124, row_ror:8=0x128 (rotations complete a commutative butterfly)
template <int CTRL>
__device__ __forceinline__ float dppmov(float x) {
  return __builtin_bit_cast(float,
      __builtin_amdgcn_mov_dpp(__builtin_bit_cast(int, x), CTRL, 0xf, 0xf, true));
}

__device__ __forceinline__ float rmax16(float v) {
  v = fmaxf(v, dppmov<0xB1>(v));
  v = fmaxf(v, dppmov<0x4E>(v));
  v = fmaxf(v, dppmov<0x124>(v));
  v = fmaxf(v, dppmov<0x128>(v));
  return v;
}

__device__ __forceinline__ float rsum16(float v) {
  v += dppmov<0xB1>(v);
  v += dppmov<0x4E>(v);
  v += dppmov<0x124>(v);
  v += dppmov<0x128>(v);
  return v;
}

__global__ __launch_bounds__(WAVES * 32, 1)
void fa_fwd_kernel(const float* __restrict__ Q, const float* __restrict__ K,
                   const float* __restrict__ V, const float* __restrict__ Mask,
                   float* __restrict__ Out) {
  __shared__ __align__(16) __bf16 KtS[32 * KT_STRIDE];          // 8704 B
  __shared__ __align__(16) __bf16 VtS[DIM * VT_STRIDE];         // 10240 B, V^T
  __shared__ __align__(16) __bf16 PtS[WAVES][16 * PT_STRIDE];   // 5120 B

  const int tid  = threadIdx.x;
  const int lane = tid & 31;
  const int wave = tid >> 5;
  const int m    = lane & 15;   // A-row / B-column index
  const int h    = lane >> 4;   // lane half

  const int qtiles = SEQ / (16 * WAVES);          // 32 q-tiles per batch-head
  const int b  = blockIdx.x / qtiles;
  const int qt = blockIdx.x % qtiles;
  const int qb = qt * (16 * WAVES) + wave * 16;   // this wave's first q row

  const float* Qb = Q    + ((size_t)b * SEQ + qb) * DIM;
  const float* Kb = K    + (size_t)b * SEQ * DIM;
  const float* Vb = V    + (size_t)b * SEQ * DIM;
  const float* Mb = Mask + (size_t)b * SEQ * SEQ;

  // ---- Q fragments: 16 rows x 128 cols as 4 bf16 A-fragments (VGPR-resident)
  v16bf qf[4];
  {
    const float* qrow = Qb + (size_t)m * DIM;
    #pragma unroll
    for (int kk = 0; kk < 4; ++kk) qf[kk] = load_fragA_f32(qrow + kk * 32, h);
  }

  // ---- online-softmax state (row r = g + 8h lives at per-lane index g)
  v8f o[8];
  #pragma unroll
  for (int c = 0; c < 8; ++c) o[c] = (v8f){0.f,0.f,0.f,0.f,0.f,0.f,0.f,0.f};
  float mx[8], ls[8];
  #pragma unroll
  for (int g = 0; g < 8; ++g) { mx[g] = -INFINITY; ls[g] = 0.f; }

  // staging assignments (block-wide, 128 threads)
  const int kr = tid >> 2;          // K stage: row 0..31
  const int kq = tid & 3;           // K stage: 32-col quadrant

  for (int j = 0; j < SEQ; j += BN) {
    __syncthreads();   // previous iteration done reading Kt/Vt

    // ---- stage K tile (32 x 128) as bf16, row-major, padded stride
    {
      const float* ksrc = Kb + (size_t)(j + kr) * DIM + kq * 32;
      __bf16* kdst = KtS + kr * KT_STRIDE + kq * 32;
      #pragma unroll
      for (int i = 0; i < 4; ++i) {
        float4 a = *(const float4*)(ksrc + 8 * i);
        float4 c = *(const float4*)(ksrc + 8 * i + 4);
        *(v8bf*)(kdst + 8 * i) = cvt8(a, c);
      }
    }
    // ---- stage V tile transposed: VtS[d][r] = V[j+r, d], packed pair stores
    {
      const float* vsrc = Vb + (size_t)j * DIM + tid;   // column d = tid
      __bf16* vdst = VtS + tid * VT_STRIDE;
      #pragma unroll
      for (int i = 0; i < 16; ++i) {
        float a = vsrc[(size_t)(2 * i) * DIM];
        float c = vsrc[(size_t)(2 * i + 1) * DIM];
        v2bf w; w[0] = (bf16)a; w[1] = (bf16)c;
        *(v2bf*)(vdst + 2 * i) = w;
      }
    }
    __syncthreads();   // staged tiles visible

    // ---- S tile = Q @ K^T : 16 x 32, two f32 accumulators
    v8f s0 = (v8f){0.f,0.f,0.f,0.f,0.f,0.f,0.f,0.f};
    v8f s1 = (v8f){0.f,0.f,0.f,0.f,0.f,0.f,0.f,0.f};
    #pragma unroll
    for (int kk = 0; kk < 4; ++kk) {
      // B-fragment: lane = column n (=m), half h covers k = 8h..8h+7, 16+8h..16+8h+7
      const __bf16* kp0 = KtS + m * KT_STRIDE + kk * 32;
      v16bf b0 = ldsfragA(kp0, h);
      v16bf b1 = ldsfragA(kp0 + 16 * KT_STRIDE, h);
      s0 = __builtin_amdgcn_wmma_f32_16x16x32_bf16(false, qf[kk], false, b0,
                                                   (short)0, s0, false, false);
      s1 = __builtin_amdgcn_wmma_f32_16x16x32_bf16(false, qf[kk], false, b1,
                                                   (short)0, s1, false, false);
    }

    // ---- additive mask, streamed non-temporally (single use; keep K/V in L2)
    #pragma unroll
    for (int g = 0; g < 8; ++g) {
      const size_t row = (size_t)(qb + g + 8 * h);
      s0[g] += __builtin_nontemporal_load(&Mb[row * SEQ + j + m]);
      s1[g] += __builtin_nontemporal_load(&Mb[row * SEQ + j + 16 + m]);
    }

    // ---- online softmax update (pure-VALU DPP butterflies)
    float p0v[8], p1v[8], sc[8];
    #pragma unroll
    for (int g = 0; g < 8; ++g) {
      float r = rmax16(fmaxf(s0[g], s1[g]));
      float mn = fmaxf(mx[g], r);
      sc[g] = __expf(mx[g] - mn);
      p0v[g] = __expf(s0[g] - mn);
      p1v[g] = __expf(s1[g] - mn);
      float t = rsum16(p0v[g] + p1v[g]);
      ls[g] = ls[g] * sc[g] + t;
      mx[g] = mn;
    }
    #pragma unroll
    for (int c = 0; c < 8; ++c) {
      #pragma unroll
      for (int g = 0; g < 8; ++g) o[c][g] *= sc[g];
    }

    // ---- P: D-layout -> A-layout via per-wave LDS round trip (bf16).
    // Per-wave region + same-wave DS ops execute in order (ISA 7.3), so a
    // compiler-ordering fence is sufficient; no block barrier needed here.
    __bf16* Pw = &PtS[wave][0];
    #pragma unroll
    for (int g = 0; g < 8; ++g) {
      Pw[(g + 8 * h) * PT_STRIDE + m]      = (bf16)p0v[g];
      Pw[(g + 8 * h) * PT_STRIDE + 16 + m] = (bf16)p1v[g];
    }
    asm volatile("" ::: "memory");   // keep DS store->load issue order
    v16bf pf = ldsfragA(Pw + m * PT_STRIDE, h);

    // ---- O += P @ V : B-fragment of V from transposed LDS tile
    #pragma unroll
    for (int c = 0; c < 8; ++c) {
      // lane = output column d = c*16 + m, half h covers kv rows 8h..8h+7, 16+8h..16+8h+7
      v16bf vf = ldsfragA(VtS + (c * 16 + m) * VT_STRIDE, h);
      o[c] = __builtin_amdgcn_wmma_f32_16x16x32_bf16(false, pf, false, vf,
                                                     (short)0, o[c], false, false);
    }
  }

  // ---- normalize and write out (fp32, non-temporal: written once)
  float* Ob = Out + ((size_t)b * SEQ + qb) * DIM;
  #pragma unroll
  for (int g = 0; g < 8; ++g) {
    float inv = 1.0f / ls[g];
    #pragma unroll
    for (int c = 0; c < 8; ++c) {
      __builtin_nontemporal_store(o[c][g] * inv,
                                  &Ob[(size_t)(g + 8 * h) * DIM + c * 16 + m]);
    }
  }
}

extern "C" void kernel_launch(void* const* d_in, const int* in_sizes, int n_in,
                              void* d_out, int out_size, void* d_ws, size_t ws_size,
                              hipStream_t stream) {
  (void)in_sizes; (void)n_in; (void)d_ws; (void)ws_size; (void)out_size;
  const float* q    = (const float*)d_in[0];
  const float* k    = (const float*)d_in[1];
  const float* v    = (const float*)d_in[2];
  const float* mask = (const float*)d_in[3];
  float* out = (float*)d_out;

  dim3 grid(NBH * (SEQ / (16 * WAVES)));   // 16 * 32 = 512 workgroups
  dim3 block(WAVES * 32);                  // 128 threads = 4 wave32
  fa_fwd_kernel<<<grid, block, 0, stream>>>(q, k, v, mask, out);
}
